// MambaBlock_206158430231
// MI455X (gfx1250) — matrix operand order repaired and checked by reading, not compiled, not run on hardware
//
#include <hip/hip_runtime.h>
#include <hip/hip_bf16.h>

// ---------------------------------------------------------------------------
// Mamba block for MI455X (gfx1250, wave32, WMMA).
// GEMMs run on v_wmma_f32_16x16x32_bf16 (bf16 in, f32 accumulate), fully
// compile-time specialized so accumulators stay pinned in VGPRs (no spills).
// ---------------------------------------------------------------------------

#define B_SZ     2
#define T_LEN    1024
#define D_MODEL  1024
#define D_STATE  16
#define D_CONV   4
#define D_INNER  2048
#define DT_RANK  64
#define ROWS     (B_SZ * T_LEN)          // 2048 (b,t) rows
#define XDBC_N   (DT_RANK + 2 * D_STATE) // 96

typedef __bf16 bf16_t;
typedef __attribute__((ext_vector_type(16))) __bf16       v16bf;
typedef __attribute__((ext_vector_type(8)))  float        v8f;
typedef __attribute__((ext_vector_type(4)))  unsigned int u32x4;

union Frag32B { u32x4 u[2]; v16bf v; };

static __device__ __forceinline__ bf16_t f2bf(float f) {
    union { float f; unsigned int u; } in; in.f = f;
    unsigned int r = (in.u + 0x7FFFu + ((in.u >> 16) & 1u)) >> 16;  // RNE
    union { unsigned short s; bf16_t b; } out; out.s = (unsigned short)r;
    return out.b;
}

static __device__ __forceinline__ float silu(float v) {
    return v * (1.0f / (1.0f + __expf(-v)));
}

// ---------------------------------------------------------------------------
// f32 -> bf16 convert (weights, activations)
// ---------------------------------------------------------------------------
__global__ __launch_bounds__(256) void k_convert_bf16(
        const float* __restrict__ src, bf16_t* __restrict__ dst, int n) {
    int i = blockIdx.x * blockDim.x + threadIdx.x;
    if (i < n) dst[i] = f2bf(src[i]);
}

// ---------------------------------------------------------------------------
// RMSNorm: one block per (b,t) row of 1024; write bf16 normalized row.
// ---------------------------------------------------------------------------
__global__ __launch_bounds__(256) void k_rmsnorm(
        const float* __restrict__ x, const float* __restrict__ norm_w,
        bf16_t* __restrict__ xn_bf) {
    __shared__ float ssum[256];
    const int row = blockIdx.x;
    const int tid = threadIdx.x;
    const float* xr = x + (long)row * D_MODEL;

    float v0 = xr[tid * 4 + 0], v1 = xr[tid * 4 + 1];
    float v2 = xr[tid * 4 + 2], v3 = xr[tid * 4 + 3];
    ssum[tid] = v0 * v0 + v1 * v1 + v2 * v2 + v3 * v3;
    __syncthreads();
    for (int s = 128; s > 0; s >>= 1) {
        if (tid < s) ssum[tid] += ssum[tid + s];
        __syncthreads();
    }
    const float rs = rsqrtf(ssum[0] * (1.0f / D_MODEL) + 1.1920929e-7f);
    bf16_t* o = xn_bf + (long)row * D_MODEL;
    o[tid * 4 + 0] = f2bf(v0 * rs * norm_w[tid * 4 + 0]);
    o[tid * 4 + 1] = f2bf(v1 * rs * norm_w[tid * 4 + 1]);
    o[tid * 4 + 2] = f2bf(v2 * rs * norm_w[tid * 4 + 2]);
    o[tid * 4 + 3] = f2bf(v3 * rs * norm_w[tid * 4 + 3]);
}

// ---------------------------------------------------------------------------
// WMMA GEMM, compile-time specialized: C[ROWS,N] = A[ROWS,K] @ W[N,K]^T.
// Block = 8 waves; wave computes 16(M) x (16*JT)(N); grid = (N/(16*JT), ROWS/128).
// MODE 0: store raw  MODE 1: softplus(acc + extra[n])  MODE 2: acc + extra[r*N+n]
// VGPR layouts per CDNA5 ISA 7.12.2:
//   A (16x32 bf16): lane<16 -> M=lane, K = {k..k+7, k+16..k+23}
//                   lane>=16 -> M=lane-16, K = {k+8..k+15, k+24..k+31}
//   B (32x16 bf16): lane<16 -> N=lane, K={k..k+7,k+16..k+23} ; lane>=16 -> N=lane-16, K={k+8..k+15,k+24..k+31}
//   C/D f32: VGPR v -> row m0 + v + 8*(lane>=16), col n0 + (lane&15)
// ---------------------------------------------------------------------------
template <int N, int K, int MODE, int JT>
__global__ __launch_bounds__(256) void k_gemm(
        const bf16_t* __restrict__ A, const bf16_t* __restrict__ W,
        float* __restrict__ C, const float* __restrict__ extra) {
    const int lane    = threadIdx.x & 31;
    const int wave    = threadIdx.x >> 5;
    const int m0      = blockIdx.y * 128 + wave * 16;
    const int n0      = blockIdx.x * (16 * JT);
    const int lane_lo = lane & 15;

    v8f acc[JT] = {};

    const bf16_t* aptr = A + (long)(m0 + lane_lo) * K + ((lane < 16) ? 0 : 8);
    const bf16_t* wptr[JT];
#pragma unroll
    for (int j = 0; j < JT; ++j)
        wptr[j] = W + (long)(n0 + j * 16 + lane_lo) * K + ((lane < 16) ? 0 : 8);

#pragma unroll 2
    for (int k = 0; k < K; k += 32) {
        Frag32B fa;
        fa.u[0] = *reinterpret_cast<const u32x4*>(aptr + k);
        fa.u[1] = *reinterpret_cast<const u32x4*>(aptr + k + 16);
        const v16bf av = fa.v;
#pragma unroll
        for (int j = 0; j < JT; ++j) {
            Frag32B fb;
            fb.u[0] = *reinterpret_cast<const u32x4*>(wptr[j] + k);
            fb.u[1] = *reinterpret_cast<const u32x4*>(wptr[j] + k + 16);
            acc[j] = __builtin_amdgcn_wmma_f32_16x16x32_bf16(
                false, av, false, fb.v, (short)0, acc[j], false, false);
        }
    }

    const int rofs = (lane >= 16) ? 8 : 0;
#pragma unroll
    for (int j = 0; j < JT; ++j) {
        const int col = n0 + j * 16 + lane_lo;
#pragma unroll
        for (int v = 0; v < 8; ++v) {
            const int  row = m0 + v + rofs;
            const long idx = (long)row * N + col;
            float val = acc[j][v];
            if (MODE == 1) {
                const float t = val + extra[col];
                val = (t > 20.0f) ? t : log1pf(__expf(t));
            } else if (MODE == 2) {
                val += extra[idx];
            }
            C[idx] = val;
        }
    }
}

// ---------------------------------------------------------------------------
// Causal depthwise conv (D_CONV=4) + bias + SiLU.  x_s lives in xz[..., :2048].
// Writes u in both f32 (scan) and bf16 (x_proj GEMM A-operand).
// ---------------------------------------------------------------------------
__global__ __launch_bounds__(256) void k_conv_silu(
        const float* __restrict__ xz, const float* __restrict__ conv_w,
        const float* __restrict__ conv_b, float* __restrict__ u_f32,
        bf16_t* __restrict__ u_bf) {
    const int idx = blockIdx.x * blockDim.x + threadIdx.x;
    const int i = idx % D_INNER;
    const int t = (idx / D_INNER) % T_LEN;
    const int b = idx / (D_INNER * T_LEN);

    float acc = conv_b[i];
#pragma unroll
    for (int j = 0; j < D_CONV; ++j) {
        const int tj = t - (D_CONV - 1) + j;
        if (tj >= 0)
            acc += conv_w[i * D_CONV + j] *
                   xz[((long)(b * T_LEN + tj)) * (2 * D_INNER) + i];
    }
    const float v = silu(acc);
    const long o = (long)(b * T_LEN + t) * D_INNER + i;
    u_f32[o] = v;
    u_bf[o]  = f2bf(v);
}

// ---------------------------------------------------------------------------
// Extract dt_raw = x_dbc[..., :DT_RANK] as bf16 (A-operand of dt GEMM).
// ---------------------------------------------------------------------------
__global__ __launch_bounds__(256) void k_dt_extract(
        const float* __restrict__ x_dbc, bf16_t* __restrict__ dt_bf) {
    const int idx = blockIdx.x * blockDim.x + threadIdx.x;
    const int row = idx / DT_RANK;
    const int r   = idx % DT_RANK;
    dt_bf[idx] = f2bf(x_dbc[(long)row * XDBC_N + r]);
}

// ---------------------------------------------------------------------------
// Selective scan, fused with D-skip + SiLU(z) gating -> bf16 gated output.
// One thread per (b, channel i); 16-state recurrence in registers.
// B_ssm/C_ssm tiles (64 timesteps x 32 floats) staged through LDS.
// ---------------------------------------------------------------------------
__global__ __launch_bounds__(256) void k_scan(
        const float* __restrict__ delta,   // [ROWS, D_INNER]
        const float* __restrict__ u_f32,   // [ROWS, D_INNER]
        const float* __restrict__ x_dbc,   // [ROWS, 96]
        const float* __restrict__ xz,      // [ROWS, 4096] (z at +2048)
        const float* __restrict__ A_log,   // [D_INNER, 16]
        const float* __restrict__ D_param, // [D_INNER]
        bf16_t* __restrict__ yg_bf) {      // [ROWS, D_INNER]
    __shared__ float sBC[64 * 32];
    const int b = blockIdx.x / (D_INNER / 256);
    const int i = (blockIdx.x % (D_INNER / 256)) * 256 + threadIdx.x;

    float An[D_STATE], h[D_STATE];
#pragma unroll
    for (int s = 0; s < D_STATE; ++s) {
        An[s] = -__expf(A_log[i * D_STATE + s]);
        h[s]  = 0.0f;
    }
    const float Dp = D_param[i];

    for (int t0 = 0; t0 < T_LEN; t0 += 64) {
#pragma unroll
        for (int r = 0; r < 8; ++r) {
            const int e  = threadIdx.x + 256 * r;
            const int tt = e >> 5, c = e & 31;
            sBC[e] = x_dbc[((long)(b * T_LEN + t0 + tt)) * XDBC_N + DT_RANK + c];
        }
        __syncthreads();

        for (int tt = 0; tt < 64; ++tt) {
            const long row = (long)(b * T_LEN + t0 + tt);
            const float d  = delta[row * D_INNER + i];
            const float uv = u_f32[row * D_INNER + i];
            const float du = d * uv;
            const float* bc = &sBC[tt * 32];
            float y = 0.0f;
#pragma unroll
            for (int s = 0; s < D_STATE; ++s) {
                h[s] = __expf(d * An[s]) * h[s] + du * bc[s];
                y += h[s] * bc[D_STATE + s];
            }
            const float z = xz[row * (2 * D_INNER) + D_INNER + i];
            yg_bf[row * D_INNER + i] = f2bf((y + uv * Dp) * silu(z));
        }
        __syncthreads();
    }
}

// ---------------------------------------------------------------------------
// Host-side orchestration
// ---------------------------------------------------------------------------
extern "C" void kernel_launch(void* const* d_in, const int* in_sizes, int n_in,
                              void* d_out, int out_size, void* d_ws, size_t ws_size,
                              hipStream_t stream) {
    (void)in_sizes; (void)n_in; (void)out_size; (void)ws_size;
    const float* x          = (const float*)d_in[0];
    const float* norm_w     = (const float*)d_in[1];
    const float* in_proj_w  = (const float*)d_in[2];
    const float* conv_w     = (const float*)d_in[3];
    const float* conv_b     = (const float*)d_in[4];
    const float* x_proj_w   = (const float*)d_in[5];
    const float* dt_proj_w  = (const float*)d_in[6];
    const float* dt_proj_b  = (const float*)d_in[7];
    const float* A_log      = (const float*)d_in[8];
    const float* D_param    = (const float*)d_in[9];
    const float* out_proj_w = (const float*)d_in[10];
    float* out = (float*)d_out;

    // Workspace carve (256B aligned regions).
    char* p = (char*)d_ws;
    auto carve = [&](size_t bytes) { char* r = p; p += (bytes + 255) & ~size_t(255); return r; };
    bf16_t* xn_bf   = (bf16_t*)carve((size_t)ROWS * D_MODEL * 2);
    bf16_t* winA    = (bf16_t*)carve((size_t)2 * D_INNER * D_MODEL * 2);
    float*  xz      = (float*) carve((size_t)ROWS * 2 * D_INNER * 4);
    float*  u_f32   = (float*) carve((size_t)ROWS * D_INNER * 4);
    bf16_t* u_bf    = (bf16_t*)carve((size_t)ROWS * D_INNER * 2);
    bf16_t* wxp     = (bf16_t*)carve((size_t)XDBC_N * D_INNER * 2);
    float*  x_dbc   = (float*) carve((size_t)ROWS * XDBC_N * 4);
    bf16_t* dt_bf   = (bf16_t*)carve((size_t)ROWS * DT_RANK * 2);
    bf16_t* wdt     = (bf16_t*)carve((size_t)D_INNER * DT_RANK * 2);
    float*  delta   = (float*) carve((size_t)ROWS * D_INNER * 4);
    bf16_t* yg_bf   = (bf16_t*)carve((size_t)ROWS * D_INNER * 2);
    bf16_t* wout    = (bf16_t*)carve((size_t)D_MODEL * D_INNER * 2);

    const int CVT = 256;
    k_convert_bf16<<<(2 * D_INNER * D_MODEL) / CVT, CVT, 0, stream>>>(in_proj_w, winA, 2 * D_INNER * D_MODEL);
    k_convert_bf16<<<(XDBC_N * D_INNER) / CVT, CVT, 0, stream>>>(x_proj_w, wxp, XDBC_N * D_INNER);
    k_convert_bf16<<<(D_INNER * DT_RANK) / CVT, CVT, 0, stream>>>(dt_proj_w, wdt, D_INNER * DT_RANK);
    k_convert_bf16<<<(D_MODEL * D_INNER) / CVT, CVT, 0, stream>>>(out_proj_w, wout, D_MODEL * D_INNER);

    // 1) RMSNorm -> bf16
    k_rmsnorm<<<ROWS, 256, 0, stream>>>(x, norm_w, xn_bf);

    // 2) in_proj: [2048,1024] x [4096,1024]^T -> xz [2048,4096]
    k_gemm<2 * D_INNER, D_MODEL, 0, 4>
        <<<dim3((2 * D_INNER) / 64, ROWS / 128), 256, 0, stream>>>(xn_bf, winA, xz, nullptr);

    // 3) causal depthwise conv + SiLU -> u
    k_conv_silu<<<(B_SZ * T_LEN * D_INNER) / 256, 256, 0, stream>>>(
        xz, conv_w, conv_b, u_f32, u_bf);

    // 4) x_proj: [2048,2048] x [96,2048]^T -> x_dbc [2048,96]  (96 = 3 * 32)
    k_gemm<XDBC_N, D_INNER, 0, 2>
        <<<dim3(XDBC_N / 32, ROWS / 128), 256, 0, stream>>>(u_bf, wxp, x_dbc, nullptr);

    // 5) dt slice -> bf16
    k_dt_extract<<<(ROWS * DT_RANK) / 256, 256, 0, stream>>>(x_dbc, dt_bf);

    // 6) delta = softplus(dt_raw @ dt_proj_w^T + dt_proj_b)
    k_gemm<D_INNER, DT_RANK, 1, 4>
        <<<dim3(D_INNER / 64, ROWS / 128), 256, 0, stream>>>(dt_bf, wdt, delta, dt_proj_b);

    // 7) selective scan fused with D-skip + SiLU(z) gating -> yg_bf
    k_scan<<<B_SZ * (D_INNER / 256), 256, 0, stream>>>(
        delta, u_f32, x_dbc, xz, A_log, D_param, yg_bf);

    // 8) out_proj + residual: [2048,2048] x [1024,2048]^T + x -> out
    k_gemm<D_MODEL, D_INNER, 2, 4>
        <<<dim3(D_MODEL / 64, ROWS / 128), 256, 0, stream>>>(yg_bf, wout, out, x);
}
